// BahdanauAttention_63462436766285
// MI455X (gfx1250) — matrix operand order repaired and checked by reading, not compiled, not run on hardware
//
#include <hip/hip_runtime.h>
#include <hip/hip_bf16.h>

// Bahdanau attention, MI455X (gfx1250) plan:
//   - dominant GEMM (65536x1024x1024) done with v_wmma_f32_16x16x32_bf16
//   - tanh + v-dot fused into GEMM epilogue (saves 512MB of HBM traffic)
//   - encoder_outputs read exactly twice (GEMM pass + context pass) ~ 22us HBM floor
// B=32, S=2048, H=1024

#define BB 32
#define SS 2048
#define HH 1024
#define LDA 1032  // padded bf16 LDS row stride (breaks 16-way bank conflicts)

typedef __attribute__((ext_vector_type(16))) __bf16          v16bf;
typedef __attribute__((ext_vector_type(8)))  float           v8f;
typedef __attribute__((ext_vector_type(8)))  unsigned short  u16x8;

union Frag {
  u16x8 h[2];
  v16bf v;
};

__device__ __forceinline__ unsigned short f2bf(float f) {
  unsigned u = __builtin_bit_cast(unsigned, f);
  u += 0x7fffu + ((u >> 16) & 1u);   // round-to-nearest-even
  return (unsigned short)(u >> 16);
}

__device__ __forceinline__ float ftanh(float x) {
  float e = __expf(2.0f * x);        // hardware v_exp_f32 path
  return 1.0f - 2.0f / (e + 1.0f);
}

// ---------------------------------------------------------------- kernel 1
// Convert W_c and W_h to bf16 in workspace.
__global__ void __launch_bounds__(256)
k_convert(const float* __restrict__ Wc, const float* __restrict__ Wh,
          unsigned short* __restrict__ wcb, unsigned short* __restrict__ whb) {
  int idx = blockIdx.x * 256 + threadIdx.x;   // grid covers H*H exactly
  wcb[idx] = f2bf(Wc[idx]);
  whb[idx] = f2bf(Wh[idx]);
}

// ---------------------------------------------------------------- kernel 2
// hpc[b,o] = sum_h hidden[b,h]*W_h[o,h] + b_h[o] + b_c[o]   (WMMA, M=32)
__global__ void __launch_bounds__(256)
k_hproj(const float* __restrict__ hidden, const unsigned short* __restrict__ whb,
        const float* __restrict__ b_h, const float* __restrict__ b_c,
        float* __restrict__ hpc) {
  __shared__ unsigned short sA[16 * LDA];
  const int tid = threadIdx.x;
  const int mb  = blockIdx.x;                 // 2 blocks x 16 batches
  const float* hb = hidden + mb * 16 * HH;
  for (int t = tid; t < 16 * HH; t += 256)
    sA[(t >> 10) * LDA + (t & 1023)] = f2bf(hb[t]);
  __syncthreads();

  const int w    = tid >> 5, lane = tid & 31;
  const int col  = lane & 15;
  const int hi   = (lane >> 4) & 1;
  const int kA   = hi * 8, kB = hi * 8;
  const int obase = w * 128;                  // 8 waves x 8 o-tiles = 1024 cols

  v8f acc[8];
#pragma unroll
  for (int j = 0; j < 8; ++j) { v8f z = {}; acc[j] = z; }

  for (int k0 = 0; k0 < HH; k0 += 32) {
    Frag a;
    const unsigned short* ap = &sA[(lane & 15) * LDA + k0 + kA];
    a.h[0] = *(const u16x8*)(ap);
    a.h[1] = *(const u16x8*)(ap + 16);
#pragma unroll
    for (int j = 0; j < 8; ++j) {
      Frag bf;
      const unsigned short* bp = whb + (obase + j * 16 + col) * HH + k0 + kB;
      bf.h[0] = *(const u16x8*)(bp);
      bf.h[1] = *(const u16x8*)(bp + 16);
      acc[j] = __builtin_amdgcn_wmma_f32_16x16x32_bf16(
          false, a.v, false, bf.v, (short)0, acc[j], false, false);
    }
  }

#pragma unroll
  for (int j = 0; j < 8; ++j) {
    int o = obase + j * 16 + col;
    float bias = b_h[o] + b_c[o];
#pragma unroll
    for (int i = 0; i < 8; ++i) {
      int m = mb * 16 + hi * 8 + i;           // D layout: vgpr i -> row i (+8 hi half)
      hpc[m * HH + o] = acc[j][i] + bias;
    }
  }
}

// ---------------------------------------------------------------- kernel 3
// scores[b,s] = sum_o v[o] * tanh(hpc[b,o] + sum_h enc[b,s,h]*W_c[o,h])
// Block: 64 s-rows (bf16 A in dynamic LDS). Wave: 128 o-cols.
// B-fragments kept in registers per K-step, reused over 2 A-subtiles.
__global__ void __launch_bounds__(256)
k_scores(const float* __restrict__ enc, const unsigned short* __restrict__ wcb,
         const float* __restrict__ hpc, const float* __restrict__ vvec,
         float* __restrict__ scores) {
  extern __shared__ unsigned char smem[];
  unsigned short* sA  = (unsigned short*)smem;           // 64 * LDA bf16
  float*          sHP = (float*)(sA + 64 * LDA);         // HH
  float*          sV  = sHP + HH;                        // HH
  float*          sSc = sV + HH;                         // 64

  const int tid = threadIdx.x;
  const int b   = blockIdx.x >> 5;            // 32 s-tiles of 64 per batch
  const int s0  = (blockIdx.x & 31) * 64;

  const float* encB = enc + ((size_t)b * SS + s0) * HH;
  for (int t = tid; t < 64 * HH; t += 256)
    sA[(t >> 10) * LDA + (t & 1023)] = f2bf(encB[t]);
  for (int t = tid; t < HH; t += 256) { sHP[t] = hpc[b * HH + t]; sV[t] = vvec[t]; }
  if (tid < 64) sSc[tid] = 0.0f;
  __syncthreads();

  const int w     = tid >> 5, lane = tid & 31;
  const int col   = lane & 15;
  const int hi    = (lane >> 4) & 1;
  const int kA    = hi * 8, kB = hi * 8;
  const int obase = w * 128;

#pragma unroll
  for (int mtp = 0; mtp < 2; ++mtp) {
    v8f acc0[8], acc1[8];
#pragma unroll
    for (int j = 0; j < 8; ++j) { v8f z = {}; acc0[j] = z; acc1[j] = z; }
    const int r0 = (mtp * 2 + 0) * 16 + (lane & 15);
    const int r1 = (mtp * 2 + 1) * 16 + (lane & 15);

    for (int k0 = 0; k0 < HH; k0 += 32) {
      Frag bfr[8];
#pragma unroll
      for (int j = 0; j < 8; ++j) {
        const unsigned short* bp = wcb + (obase + j * 16 + col) * HH + k0 + kB;
        bfr[j].h[0] = *(const u16x8*)(bp);
        bfr[j].h[1] = *(const u16x8*)(bp + 16);
      }
      Frag a0, a1;
      const unsigned short* ap0 = &sA[r0 * LDA + k0 + kA];
      const unsigned short* ap1 = &sA[r1 * LDA + k0 + kA];
      a0.h[0] = *(const u16x8*)(ap0); a0.h[1] = *(const u16x8*)(ap0 + 16);
      a1.h[0] = *(const u16x8*)(ap1); a1.h[1] = *(const u16x8*)(ap1 + 16);
#pragma unroll
      for (int j = 0; j < 8; ++j)
        acc0[j] = __builtin_amdgcn_wmma_f32_16x16x32_bf16(
            false, a0.v, false, bfr[j].v, (short)0, acc0[j], false, false);
#pragma unroll
      for (int j = 0; j < 8; ++j)
        acc1[j] = __builtin_amdgcn_wmma_f32_16x16x32_bf16(
            false, a1.v, false, bfr[j].v, (short)0, acc1[j], false, false);
    }

    float p0[8], p1[8];
#pragma unroll
    for (int i = 0; i < 8; ++i) { p0[i] = 0.0f; p1[i] = 0.0f; }
#pragma unroll
    for (int j = 0; j < 8; ++j) {
      int o = obase + j * 16 + col;
      float vo = sV[o], hp = sHP[o];
#pragma unroll
      for (int i = 0; i < 8; ++i) {
        p0[i] += vo * ftanh(hp + acc0[j][i]);
        p1[i] += vo * ftanh(hp + acc1[j][i]);
      }
    }
#pragma unroll
    for (int i = 0; i < 8; ++i) {
      atomicAdd(&sSc[(mtp * 2 + 0) * 16 + hi * 8 + i], p0[i]);  // ds_add_f32
      atomicAdd(&sSc[(mtp * 2 + 1) * 16 + hi * 8 + i], p1[i]);
    }
  }
  __syncthreads();
  if (tid < 64) scores[(size_t)b * SS + s0 + tid] = sSc[tid];
}

// ---------------------------------------------------------------- kernel 4
__global__ void __launch_bounds__(256)
k_softmax(const float* __restrict__ scores, float* __restrict__ attn) {
  __shared__ float red[256];
  const int b = blockIdx.x, tid = threadIdx.x;
  float loc[8];
  float m = -3.4e38f;
#pragma unroll
  for (int q = 0; q < 8; ++q) {
    loc[q] = scores[(size_t)b * SS + q * 256 + tid];
    m = fmaxf(m, loc[q]);
  }
  red[tid] = m; __syncthreads();
  for (int s = 128; s > 0; s >>= 1) {
    if (tid < s) red[tid] = fmaxf(red[tid], red[tid + s]);
    __syncthreads();
  }
  m = red[0]; __syncthreads();
  float sum = 0.0f;
#pragma unroll
  for (int q = 0; q < 8; ++q) { loc[q] = __expf(loc[q] - m); sum += loc[q]; }
  red[tid] = sum; __syncthreads();
  for (int s = 128; s > 0; s >>= 1) {
    if (tid < s) red[tid] += red[tid + s];
    __syncthreads();
  }
  float inv = 1.0f / red[0];
#pragma unroll
  for (int q = 0; q < 8; ++q)
    attn[(size_t)b * SS + q * 256 + tid] = loc[q] * inv;
}

// ---------------------------------------------------------------- kernel 5
// context[b,h] = sum_s attn[b,s] * enc[b,s,h]  (memory-bound, deterministic)
__global__ void __launch_bounds__(256)
k_context(const float* __restrict__ enc, const float* __restrict__ attn,
          float* __restrict__ ctx) {
  __shared__ float sAt[256];
  const int tid = threadIdx.x;
  const int b   = blockIdx.y;
  const int h   = blockIdx.x * 256 + tid;
  float acc = 0.0f;
  for (int sc = 0; sc < SS; sc += 256) {
    __syncthreads();
    sAt[tid] = attn[(size_t)b * SS + sc + tid];
    __syncthreads();
    const float* ep = enc + ((size_t)b * SS + sc) * HH + h;
    for (int si = 0; si < 256; ++si)
      acc += sAt[si] * ep[(size_t)si * HH];
  }
  ctx[b * HH + h] = acc;
}

// ---------------------------------------------------------------- launcher
extern "C" void kernel_launch(void* const* d_in, const int* in_sizes, int n_in,
                              void* d_out, int out_size, void* d_ws, size_t ws_size,
                              hipStream_t stream) {
  const float* hidden = (const float*)d_in[0];
  const float* enc    = (const float*)d_in[1];
  const float* W_h    = (const float*)d_in[2];
  const float* b_h    = (const float*)d_in[3];
  const float* W_c    = (const float*)d_in[4];
  const float* b_c    = (const float*)d_in[5];
  const float* v      = (const float*)d_in[6];

  float* out     = (float*)d_out;
  float* context = out;            // [B,H]  = 32768 floats
  float* attn    = out + BB * HH;  // [B,S]  = 65536 floats

  // workspace: Wc_bf16 | Wh_bf16 | hpc | scores  (~4.4 MB)
  unsigned short* wcb = (unsigned short*)d_ws;
  unsigned short* whb = wcb + HH * HH;
  float* hpc    = (float*)(whb + HH * HH);
  float* scores = hpc + BB * HH;

  k_convert<<<(HH * HH) / 256, 256, 0, stream>>>(W_c, W_h, wcb, whb);
  k_hproj<<<2, 256, 0, stream>>>(hidden, whb, b_h, b_c, hpc);

  const int smemC = 64 * LDA * 2 + (HH + HH + 64) * (int)sizeof(float); // 140544 B
  k_scores<<<(BB * SS) / 64, 256, smemC, stream>>>(enc, wcb, hpc, v, scores);

  k_softmax<<<BB, 256, 0, stream>>>(scores, attn);
  k_context<<<dim3(HH / 256, BB), 256, 0, stream>>>(enc, attn, context);
}
